// PairInteraction_71519795413679
// MI455X (gfx1250) — compile-verified, matrix operand order checked by reading, not run on hardware
//
#include <hip/hip_runtime.h>

typedef __attribute__((ext_vector_type(16))) _Float16 v16h;
typedef __attribute__((ext_vector_type(8)))  float    v8f;
typedef __attribute__((ext_vector_type(4)))  float    v4f;

#define N1_  1024
#define N2_  1024
#define D1_  256
#define FDIM 128
#define ODIM 64

// ---------------------------------------------------------------------------
// Kernel 1: f1 = features1 @ W1 + b1 ; f2 = features2 @ W2   (both 1024x128)
// ---------------------------------------------------------------------------
__global__ __launch_bounds__(128)
void fc_kernel(const float* __restrict__ features1,
               const float* __restrict__ features2,
               const float* __restrict__ W1,
               const float* __restrict__ b1,
               const float* __restrict__ W2,
               float* __restrict__ f1,
               float* __restrict__ f2) {
    __shared__ float srow[D1_];
    int r = blockIdx.x;
    bool first = r < N1_;
    int row = first ? r : r - N1_;
    const float* feat = (first ? features1 : features2) + (size_t)row * D1_;
    const float* W    = first ? W1 : W2;
    for (int k = threadIdx.x; k < D1_; k += 128) srow[k] = feat[k];
    __syncthreads();
    int o = threadIdx.x;
    float acc = first ? b1[o] : 0.0f;
#pragma unroll 8
    for (int k = 0; k < D1_; ++k)
        acc = fmaf(srow[k], W[(size_t)k * FDIM + o], acc);
    (first ? f1 : f2)[(size_t)row * FDIM + o] = acc;
}

// ---------------------------------------------------------------------------
// Kernel 1b: pre-pack W3 (128x64 fp32) into f16 WMMA B-fragments.
// Fragment index: ((c*4 + ot)*32 + lane)*16 + e
//   K = c*32 + (e<8 ? 0 : 16) + (lane>>4)*8 + (e&7),  o = ot*16 + (lane&15)
// ---------------------------------------------------------------------------
__global__ __launch_bounds__(256)
void pack_w3_kernel(const float* __restrict__ W3, _Float16* __restrict__ w3p) {
    int idx  = blockIdx.x * 256 + threadIdx.x;       // 0 .. 8191
    int e    = idx & 15;
    int lane = (idx >> 4) & 31;
    int ot   = (idx >> 9) & 3;
    int c    = idx >> 11;
    int k = c * 32 + ((e < 8) ? 0 : 16) + (lane >> 4) * 8 + (e & 7);
    int o = ot * 16 + (lane & 15);
    w3p[idx] = (_Float16)W3[(size_t)k * ODIM + o];
}

// ---------------------------------------------------------------------------
// Kernel 2: out[m, n0:n0+16, 0:64] = relu(f1[m]+f2[n]) @ W3 + b3
// One wave per (m, 16-wide n tile); 8 waves per block.
//  - W3 fragments staged once per block in LDS (all waves share them)
//  - 4 K-chunks x 4 o-tiles = 16 v_wmma_f32_16x16x32_f16 per wave
//  - Epilogue transposes the 16x64 tile through LDS so stores are 8
//    contiguous b128 per lane; stores are NON-TEMPORAL so the 256 MB
//    streaming output doesn't evict the L2-resident f1/f2/W3 working set
// ---------------------------------------------------------------------------
__global__ __launch_bounds__(256)
void pair_kernel(const float* __restrict__ f1,
                 const float* __restrict__ f2,
                 const _Float16* __restrict__ w3p,
                 const float* __restrict__ b3,
                 float* __restrict__ out) {
    __shared__ _Float16 w3s[8192];      // 16 KB: packed W3 fragments
    __shared__ float    tbuf[8][1024];  // 32 KB: per-wave 16x64 store-transpose

    int lane  = threadIdx.x & 31;
    int wave  = threadIdx.x >> 5;
    int tile  = blockIdx.x * 8 + wave;   // 65536 tiles total
    int m     = tile >> 6;               // 0..1023
    int n0    = (tile & 63) << 4;        // 0,16,...,1008
    int khalf = lane >> 4;               // 0|1 : which K half this lane holds
    int col   = lane & 15;               // A row (n offset) == C/D column

    // Stage packed W3 into LDS (1024 float4 = 16 KB, 256 threads cooperative)
    {
        const v4f* src = (const v4f*)w3p;
        v4f*       dst = (v4f*)w3s;
#pragma unroll
        for (int i = 0; i < 4; ++i)
            dst[i * 256 + threadIdx.x] = src[i * 256 + threadIdx.x];
    }
    __syncthreads();

    const float4* f1v = (const float4*)(f1 + (size_t)m * FDIM);
    const float4* f2v = (const float4*)(f2 + (size_t)(n0 + col) * FDIM);
    const v16h*  bmat = (const v16h*)w3s;

    v8f acc0 = {}; v8f acc1 = {}; v8f acc2 = {}; v8f acc3 = {};

#pragma unroll
    for (int c = 0; c < 4; ++c) {
        int i0 = c * 8 + khalf * 2;      // float4 index: K = c*32 + khalf*8
        int i1 = i0 + 4;                 // float4 index: K = c*32 + 16 + khalf*8
        float4 x0 = f1v[i0], x1 = f1v[i0 + 1], x2 = f1v[i1], x3 = f1v[i1 + 1];
        float4 y0 = f2v[i0], y1 = f2v[i0 + 1], y2 = f2v[i1], y3 = f2v[i1 + 1];

        v16h a;
        a[0]  = (_Float16)fmaxf(x0.x + y0.x, 0.0f);
        a[1]  = (_Float16)fmaxf(x0.y + y0.y, 0.0f);
        a[2]  = (_Float16)fmaxf(x0.z + y0.z, 0.0f);
        a[3]  = (_Float16)fmaxf(x0.w + y0.w, 0.0f);
        a[4]  = (_Float16)fmaxf(x1.x + y1.x, 0.0f);
        a[5]  = (_Float16)fmaxf(x1.y + y1.y, 0.0f);
        a[6]  = (_Float16)fmaxf(x1.z + y1.z, 0.0f);
        a[7]  = (_Float16)fmaxf(x1.w + y1.w, 0.0f);
        a[8]  = (_Float16)fmaxf(x2.x + y2.x, 0.0f);
        a[9]  = (_Float16)fmaxf(x2.y + y2.y, 0.0f);
        a[10] = (_Float16)fmaxf(x2.z + y2.z, 0.0f);
        a[11] = (_Float16)fmaxf(x2.w + y2.w, 0.0f);
        a[12] = (_Float16)fmaxf(x3.x + y3.x, 0.0f);
        a[13] = (_Float16)fmaxf(x3.y + y3.y, 0.0f);
        a[14] = (_Float16)fmaxf(x3.z + y3.z, 0.0f);
        a[15] = (_Float16)fmaxf(x3.w + y3.w, 0.0f);

        v16h bm0 = bmat[(c * 4 + 0) * 32 + lane];
        v16h bm1 = bmat[(c * 4 + 1) * 32 + lane];
        v16h bm2 = bmat[(c * 4 + 2) * 32 + lane];
        v16h bm3 = bmat[(c * 4 + 3) * 32 + lane];

        acc0 = __builtin_amdgcn_wmma_f32_16x16x32_f16(false, a, false, bm0,
                                                      (short)0, acc0, false, false);
        acc1 = __builtin_amdgcn_wmma_f32_16x16x32_f16(false, a, false, bm1,
                                                      (short)0, acc1, false, false);
        acc2 = __builtin_amdgcn_wmma_f32_16x16x32_f16(false, a, false, bm2,
                                                      (short)0, acc2, false, false);
        acc3 = __builtin_amdgcn_wmma_f32_16x16x32_f16(false, a, false, bm3,
                                                      (short)0, acc3, false, false);
    }

    float bias0 = b3[col];
    float bias1 = b3[16 + col];
    float bias2 = b3[32 + col];
    float bias3 = b3[48 + col];

    // Write D fragments (+bias) into this wave's LDS tile in row-major order:
    // element (vgpr v, lane) -> n-row = v + khalf*8, o = ot*16 + col
    float* trow = tbuf[wave];
#pragma unroll
    for (int v = 0; v < 8; ++v) {
        int r = (v + khalf * 8) * ODIM;
        trow[r +  0 + col] = acc0[v] + bias0;
        trow[r + 16 + col] = acc1[v] + bias1;
        trow[r + 32 + col] = acc2[v] + bias2;
        trow[r + 48 + col] = acc3[v] + bias3;
    }
    __syncthreads();

    // Contiguous 4 KB tile store: 8 x b128 per lane, fully coalesced,
    // non-temporal (streaming 256 MB output; keep L2 for f1/f2/W3)
    const v4f* t4   = (const v4f*)trow;
    v4f*       outv = (v4f*)(out + ((size_t)m * N2_ + (size_t)n0) * ODIM);
#pragma unroll
    for (int i = 0; i < 8; ++i)
        __builtin_nontemporal_store(t4[i * 32 + lane], &outv[i * 32 + lane]);
}

// ---------------------------------------------------------------------------
extern "C" void kernel_launch(void* const* d_in, const int* in_sizes, int n_in,
                              void* d_out, int out_size, void* d_ws, size_t ws_size,
                              hipStream_t stream) {
    const float* features1 = (const float*)d_in[0];
    const float* features2 = (const float*)d_in[1];
    const float* W1 = (const float*)d_in[2];
    const float* b1 = (const float*)d_in[3];
    const float* W2 = (const float*)d_in[4];
    const float* W3 = (const float*)d_in[5];
    const float* b3 = (const float*)d_in[6];
    float* out = (float*)d_out;

    // Workspace layout: f1 (512KB) | f2 (512KB) | packed W3 f16 (16KB)
    float*    f1  = (float*)d_ws;
    float*    f2  = f1 + (size_t)N1_ * FDIM;
    _Float16* w3p = (_Float16*)(f2 + (size_t)N2_ * FDIM);

    fc_kernel<<<N1_ + N2_, 128, 0, stream>>>(features1, features2, W1, b1, W2, f1, f2);
    pack_w3_kernel<<<8192 / 256, 256, 0, stream>>>(W3, w3p);

    // 1024 m * 64 n-tiles = 65536 waves; 8 waves (256 threads) per block
    pair_kernel<<<(N1_ * (N2_ / 16)) / 8, 256, 0, stream>>>(f1, f2, w3p, b3, out);
}
